// LocalVarianceMap_49014166782433
// MI455X (gfx1250) — compile-verified, matrix operand order checked by reading, not run on hardware
//
#include <hip/hip_runtime.h>
#include <hip/hip_bf16.h>

typedef __attribute__((ext_vector_type(16))) _Float16 v16h;
typedef __attribute__((ext_vector_type(8)))  float    v8f;

#define IMG_H 1024
#define IMG_W 1024
#define NBATCH 16
#define TILE 64
#define HALO 3
#define REG  (TILE + 2*HALO)   // 70
#define REGP 72                 // padded cols for lum arrays
#define RP   82                 // padded rows for column-major hsum (odd word pitch -> no bank conflicts)

union BU { v16h v; uint32_t u[8]; };

__global__ __launch_bounds__(256)
void local_var_map_kernel(const float* __restrict__ x, float* __restrict__ out) {
    __shared__ float    s_lum [REG][REGP];
    __shared__ float    s_lum2[REG][REGP];
    __shared__ _Float16 s_h   [TILE][RP];   // col-major: [out_col][region_row], f16
    __shared__ _Float16 s_h2  [TILE][RP];

    const int tid = threadIdx.x;
    const int x0 = blockIdx.x * TILE;
    const int y0 = blockIdx.y * TILE;
    const int bz = blockIdx.z;

    const size_t HW = (size_t)IMG_H * IMG_W;
    const float* xb = x + (size_t)bz * 3 * HW;

    // ---------------- Phase 1: load region, compute luminance into LDS ----------------
    for (int i = tid; i < REG * REG; i += 256) {
        int r = i / REG;
        int c = i - r * REG;
        int gy = y0 + r - HALO;
        int gx = x0 + c - HALO;
        float l = 0.0f;
        if (gy >= 0 && gy < IMG_H && gx >= 0 && gx < IMG_W) {
            const float* p = xb + (size_t)gy * IMG_W + gx;
            l = (p[0] + p[HW] + p[2 * HW]) * (1.0f / 3.0f);
        }
        s_lum [r][c] = l;
        s_lum2[r][c] = l * l;
    }
    __syncthreads();

    // ---------------- Phase 2: horizontal 7-tap sums -> f16 column-major ----------------
    // zero the pad rows (REG..RP-1) so WMMA K-lanes beyond the region see 0, not garbage
    for (int i = tid; i < TILE * (RP - REG); i += 256) {
        int c = i / (RP - REG);
        int r = REG + (i - c * (RP - REG));
        s_h [c][r] = (_Float16)0.0f;
        s_h2[c][r] = (_Float16)0.0f;
    }
    for (int i = tid; i < TILE * REG; i += 256) {
        int c = i & (TILE - 1);     // output column within tile (lanes -> consecutive c)
        int r = i >> 6;             // region row 0..69
        float a = 0.0f, a2 = 0.0f;
#pragma unroll
        for (int d = 0; d < 7; ++d) {
            a  += s_lum [r][c + d];
            a2 += s_lum2[r][c + d];
        }
        s_h [c][r] = (_Float16)a;
        s_h2[c][r] = (_Float16)a2;
    }
    __syncthreads();

    // ---------------- Phase 3: vertical 7-tap sum as WMMA band-matrix product ----------------
    const int lane = tid & 31;
    const int wv   = tid >> 5;           // wave id 0..7
    const int m    = lane & 15;          // matrix row handled by this lane
    const int kofs = (lane & 16);        // K-half: 0 or 16

    // A = band matrix Sv[m][k] = 1 iff k in [m, m+6]  (exact in f16)
    v16h A;
#pragma unroll
    for (int j = 0; j < 16; ++j) {
        int k = kofs + j;
        A[j] = (k >= m && k <= m + 6) ? (_Float16)1.0f : (_Float16)0.0f;
    }

    float* op = out + (size_t)bz * HW;

#pragma unroll
    for (int t = 0; t < 2; ++t) {
        int st = wv + t * 8;             // subtile 0..15
        int g  = (st >> 2) * 16;         // output-row group within tile
        int cg = (st & 3) * 16;          // output-col group within tile

        int n  = cg + m;                 // global-within-tile column for this lane
        int rb = g + kofs;               // first region row for this lane's K-half (even)

        BU b1, b2;
        const uint32_t* p1 = (const uint32_t*)(&s_h [n][rb]);   // 4B-aligned: rb even, RP even
        const uint32_t* p2 = (const uint32_t*)(&s_h2[n][rb]);
#pragma unroll
        for (int v = 0; v < 8; ++v) { b1.u[v] = p1[v]; b2.u[v] = p2[v]; }

        v8f acc1 = {};
        v8f acc2 = {};
        acc1 = __builtin_amdgcn_wmma_f32_16x16x32_f16(false, A, false, b1.v,
                                                      (short)0, acc1, false, false);
        acc2 = __builtin_amdgcn_wmma_f32_16x16x32_f16(false, A, false, b2.v,
                                                      (short)0, acc2, false, false);

        int colg    = x0 + n;
        int rowbase = y0 + g + ((lane >> 4) << 3);   // lanes 0-15: M=r, lanes 16-31: M=8+r
#pragma unroll
        for (int r = 0; r < 8; ++r) {
            float mean = acc1[r] * (1.0f / 49.0f);
            float var  = acc2[r] * (1.0f / 49.0f) - mean * mean;
            op[(size_t)(rowbase + r) * IMG_W + colg] = var;
        }
    }
}

extern "C" void kernel_launch(void* const* d_in, const int* in_sizes, int n_in,
                              void* d_out, int out_size, void* d_ws, size_t ws_size,
                              hipStream_t stream) {
    const float* x = (const float*)d_in[0];
    float* out = (float*)d_out;
    // kernel_size fixed at 7 by setup_inputs(); band matrices are specialized for k=7.
    dim3 grid(IMG_W / TILE, IMG_H / TILE, NBATCH);
    local_var_map_kernel<<<grid, 256, 0, stream>>>(x, out);
}
